// CrossAttentionFusionModule_14190571946414
// MI455X (gfx1250) — matrix operand order repaired and checked by reading, compiled, not run on hardware
//
#include <hip/hip_runtime.h>
#include <hip/hip_bf16.h>

// ---------------------------------------------------------------------------
// CrossAttentionFusion, algebraically refactored for MI455X (gfx1250, wave32):
//   G_b  = X_b Y_b^T           (256x256, K=32768)  + row sums sx, sy   [pass 1]
//   S_b  = Wq G_b Wk^T + (Wq sx) bk^T + bq (Wk sy)^T + N bq bk^T       [tiny]
//   A_b  = softmax_rows(S_b);  M_b = A_b Wv;  c_b = A_b bv             [tiny]
//   out  = M_b Y_b + c_b 1^T   (256x32768, K=256)                      [pass 2]
// HBM traffic ~536MB (memory bound, ~23us @ 23.3TB/s). Matrix math via
// v_wmma_f32_16x16x32_f16 (f16 in, f32 accumulate). Pass 1 waves are
// register-blocked 2x2 (32x32 output), pass 2 waves 4x1 (64x16 output) to cut
// L2 re-read traffic and densify the WMMA pipe.
// ---------------------------------------------------------------------------

typedef __attribute__((ext_vector_type(16))) _Float16 v16h;
typedef __attribute__((ext_vector_type(8)))  float    v8f;

#define CCH  256
#define NSP  32768      // 32*32*32
#define NB   4

// ---- WMMA operand builders (contiguous k = 16*half + s, same order in A and B) -----

// A (16x32, f16): lane l in 0..15 -> row l; elements 0..15 hold K=off+{0..15},
// off = (lane>=16 ? 16 : 0).
__device__ __forceinline__ v16h load_a_rm(const float* __restrict__ A, int lda,
                                          int row, int k0, int half) {
  const float* p = A + (long)row * lda + k0 + half * 16;
  v16h a;
#pragma unroll
  for (int t = 0; t < 8; ++t) {
    a[t]     = (_Float16)p[t];
    a[8 + t] = (_Float16)p[8 + t];
  }
  return a;
}

// B (32x16, f16) from Bt stored as [N x K] row-major (logical B = Bt^T):
// lane l -> column n=l; lanes 0..15 hold K=k0+{0..15}, lanes 16..31 K=k0+16+{0..15}.
__device__ __forceinline__ v16h load_b_tr(const float* __restrict__ Bt, int ldb,
                                          int nrow, int k0, int half) {
  const float* p = Bt + (long)nrow * ldb + k0 + half * 16;
  v16h b;
#pragma unroll
  for (int t = 0; t < 16; ++t) b[t] = (_Float16)p[t];
  return b;
}

// B (32x16, f16) from B stored as [K x N] row-major: per-lane strided, but each
// K step is a 64B contiguous run across lanes 0..15 / 16..31 (fully coalesced).
__device__ __forceinline__ v16h load_b_rm(const float* __restrict__ B, long ldb,
                                          int ncol, int k0, int half) {
  const float* p = B + (long)(k0 + half * 16) * ldb + ncol;
  v16h b;
#pragma unroll
  for (int t = 0; t < 16; ++t) b[t] = (_Float16)p[t * ldb];
  return b;
}

#define WMMA_F16(a, b, c) \
  __builtin_amdgcn_wmma_f32_16x16x32_f16(false, (a), false, (b), (short)0, (c), false, false)

// D/C f32 16x16: VGPR r -> M = r + (lane>=16 ? 8 : 0), N = lane%16.
__device__ __forceinline__ void store_tile(float* __restrict__ D, long ldd,
                                           int i0, int j0, v8f acc) {
  const int lane = threadIdx.x & 31, half = lane >> 4, l = lane & 15;
#pragma unroll
  for (int r = 0; r < 8; ++r)
    D[(long)(i0 + r + half * 8) * ldd + j0 + l] = acc[r];
}

template <bool TRANSB>
__device__ __forceinline__ v8f tile_gemm(const float* __restrict__ A, int lda, int i0,
                                         const float* __restrict__ B, long ldb, int j0,
                                         int K, v8f acc) {
  const int lane = threadIdx.x & 31, half = lane >> 4, l = lane & 15;
  for (int k0 = 0; k0 < K; k0 += 32) {
    v16h a = load_a_rm(A, lda, i0 + l, k0, half);
    v16h b = TRANSB ? load_b_tr(B, (int)ldb, j0 + l, k0, half)
                    : load_b_rm(B, ldb, j0 + l, k0, half);
    acc = WMMA_F16(a, b, acc);
  }
  return acc;
}

// ---- Pass 1: per-batch Gram G = X Y^T plus row sums.
// Wave = 32x32 output block (2x2 WMMA tiles), K split over KSLICES slices.
// Wave id decomposition puts tj2 in the low bits so the 8 waves of a block
// share the same X rows (A operand) for L2/L0 locality.
#define KSLICES 32
#define KSL     (NSP / KSLICES)   // 1024

__global__ void gram_kernel(const float* __restrict__ x, const float* __restrict__ y,
                            float* __restrict__ G, float* __restrict__ sx,
                            float* __restrict__ sy) {
  const int wid = (int)((blockIdx.x * blockDim.x + threadIdx.x) >> 5);
  const int tj2 = wid & 7, ti2 = (wid >> 3) & 7, ks = (wid >> 6) & (KSLICES - 1),
            b   = wid >> 11;
  const int lane = threadIdx.x & 31, half = lane >> 4, l = lane & 15;
  const int i0 = ti2 * 32, j0 = tj2 * 32;

  const float* __restrict__ Xb = x + (long)b * CCH * NSP;
  const float* __restrict__ Yb = y + (long)b * CCH * NSP;
  const float* a0row = Xb + (long)(i0 + l) * NSP;        // rows i0..i0+15
  const float* a1row = a0row + 16L * NSP;                // rows i0+16..i0+31
  const float* b0row = Yb + (long)(j0 + l) * NSP;
  const float* b1row = b0row + 16L * NSP;

  v8f acc00 = {}, acc01 = {}, acc10 = {}, acc11 = {};
  float sa0 = 0.f, sa1 = 0.f, sb0 = 0.f, sb1 = 0.f;

  const int kbeg = ks * KSL, kend = kbeg + KSL;
  for (int k0 = kbeg; k0 < kend; k0 += 32) {
    const int aoff = k0 + half * 16, boff = k0 + half * 16;
    __builtin_prefetch(a0row + aoff + 64, 0, 0);
    __builtin_prefetch(b0row + boff + 64, 0, 0);
    v16h a0, a1, bb0, bb1;
#pragma unroll
    for (int t = 0; t < 8; ++t) {
      float f0 = a0row[aoff + t], f1 = a0row[aoff + 8 + t];
      float g0 = a1row[aoff + t], g1 = a1row[aoff + 8 + t];
      sa0 += f0 + f1;  sa1 += g0 + g1;
      a0[t] = (_Float16)f0; a0[8 + t] = (_Float16)f1;
      a1[t] = (_Float16)g0; a1[8 + t] = (_Float16)g1;
    }
#pragma unroll
    for (int t = 0; t < 16; ++t) {
      float f = b0row[boff + t], g = b1row[boff + t];
      sb0 += f;  sb1 += g;
      bb0[t] = (_Float16)f;  bb1[t] = (_Float16)g;
    }
    acc00 = WMMA_F16(a0, bb0, acc00);
    acc01 = WMMA_F16(a0, bb1, acc01);
    acc10 = WMMA_F16(a1, bb0, acc10);
    acc11 = WMMA_F16(a1, bb1, acc11);
  }

  float* Gb = G + (long)b * CCH * CCH;
#pragma unroll
  for (int r = 0; r < 8; ++r) {
    const int ra = i0 + r + half * 8, rb = ra + 16;
    atomicAdd(&Gb[(long)ra * CCH + j0 + l],      acc00[r]);
    atomicAdd(&Gb[(long)ra * CCH + j0 + 16 + l], acc01[r]);
    atomicAdd(&Gb[(long)rb * CCH + j0 + l],      acc10[r]);
    atomicAdd(&Gb[(long)rb * CCH + j0 + 16 + l], acc11[r]);
  }

  // lanes l and l+16 each covered half the K range of their row -> combine.
  sa0 += __shfl_xor(sa0, 16, 32);
  sa1 += __shfl_xor(sa1, 16, 32);
  sb0 += __shfl_xor(sb0, 16, 32);
  sb1 += __shfl_xor(sb1, 16, 32);
  if (half == 0) {
    if (tj2 == 0) {
      atomicAdd(&sx[b * CCH + i0 + l],      sa0);
      atomicAdd(&sx[b * CCH + i0 + 16 + l], sa1);
    }
    if (ti2 == 0) {
      atomicAdd(&sy[b * CCH + j0 + l],      sb0);
      atomicAdd(&sy[b * CCH + j0 + 16 + l], sb1);
    }
  }
}

// ---- Tiny per-batch vectors u = Wq sx, w = Wk sy ---------------------------
__global__ void uv_kernel(const float* __restrict__ Wq, const float* __restrict__ Wk,
                          const float* __restrict__ sx, const float* __restrict__ sy,
                          float* __restrict__ u, float* __restrict__ w) {
  const int b = blockIdx.x, i = threadIdx.x;
  float su = 0.f, sw = 0.f;
  for (int a = 0; a < CCH; ++a) {
    su += Wq[i * CCH + a] * sx[b * CCH + a];
    sw += Wk[i * CCH + a] * sy[b * CCH + a];
  }
  u[b * CCH + i] = su;
  w[b * CCH + i] = sw;
}

// ---- Generic small 256x256x256 GEMM: D_b = A_b * B_b (B is KxN row-major) --
__global__ void smallgemm_kernel(const float* __restrict__ A, long sA,
                                 const float* __restrict__ B, long sB,
                                 float* __restrict__ D) {
  const int wid = (int)((blockIdx.x * blockDim.x + threadIdx.x) >> 5);
  const int tj = wid & 15, ti = (wid >> 4) & 15, b = wid >> 8;
  v8f acc = {};
  acc = tile_gemm<false>(A + (long)b * sA, CCH, ti * 16,
                         B + (long)b * sB, CCH, tj * 16, CCH, acc);
  store_tile(D + (long)b * CCH * CCH, CCH, ti * 16, tj * 16, acc);
}

// ---- Scores: S = T Wk^T + u bk^T + bq w^T + N bq bk^T ----------------------
__global__ void score_kernel(const float* __restrict__ T, const float* __restrict__ Wk,
                             const float* __restrict__ u, const float* __restrict__ w,
                             const float* __restrict__ bq, const float* __restrict__ bk,
                             float* __restrict__ S) {
  const int wid = (int)((blockIdx.x * blockDim.x + threadIdx.x) >> 5);
  const int tj = wid & 15, ti = (wid >> 4) & 15, b = wid >> 8;
  const int lane = threadIdx.x & 31, half = lane >> 4, l = lane & 15;
  v8f acc = {};
  acc = tile_gemm<true>(T + (long)b * CCH * CCH, CCH, ti * 16,
                        Wk, CCH, tj * 16, CCH, acc);
  const int j = tj * 16 + l;
  const float bkj = bk[j], wj = w[b * CCH + j];
#pragma unroll
  for (int r = 0; r < 8; ++r) {
    const int i = ti * 16 + r + half * 8;
    acc[r] += u[b * CCH + i] * bkj + bq[i] * (wj + (float)NSP * bkj);
  }
  store_tile(S + (long)b * CCH * CCH, CCH, ti * 16, tj * 16, acc);
}

// ---- Row softmax (in place) + c = A bv -------------------------------------
__global__ void softmax_kernel(float* __restrict__ S, const float* __restrict__ bv,
                               float* __restrict__ cvec) {
  const int row = blockIdx.x;           // b*256 + i, 1024 rows
  const int tid = threadIdx.x;          // 256 threads = 256 columns
  __shared__ float red[CCH];
  float* Srow = S + (long)row * CCH;
  const float v = Srow[tid];

  red[tid] = v; __syncthreads();
  for (int s = 128; s > 0; s >>= 1) { if (tid < s) red[tid] = fmaxf(red[tid], red[tid + s]); __syncthreads(); }
  const float m = red[0]; __syncthreads();

  const float e = __expf(v - m);
  red[tid] = e; __syncthreads();
  for (int s = 128; s > 0; s >>= 1) { if (tid < s) red[tid] += red[tid + s]; __syncthreads(); }
  const float inv = 1.0f / red[0]; __syncthreads();

  const float a = e * inv;
  Srow[tid] = a;
  red[tid] = a * bv[tid]; __syncthreads();
  for (int s = 128; s > 0; s >>= 1) { if (tid < s) red[tid] += red[tid + s]; __syncthreads(); }
  if (tid == 0) cvec[row] = red[0];
}

// ---- Pass 2: out_b = M_b Y_b + c_b 1^T  (256 x 32768, K=256) ---------------
// Wave = 64x16 output block (4 i-tiles sharing one streamed Y tile).
// ti4 sits in the low wave-id bits so the 4 waves sharing a Y tile co-reside.
__global__ void fused_kernel(const float* __restrict__ M, const float* __restrict__ cvec,
                             const float* __restrict__ y, float* __restrict__ out) {
  const int wid = (int)((blockIdx.x * blockDim.x + threadIdx.x) >> 5);
  const int ti4 = wid & 3, tn = (wid >> 2) & 2047, b = wid >> 13;
  const int lane = threadIdx.x & 31, half = lane >> 4, l = lane & 15;
  const int i0 = ti4 * 64, n0 = tn * 16;

  const float* Mb = M + (long)b * CCH * CCH;
  const float* Yb = y + (long)b * CCH * NSP;

  v8f acc0 = {}, acc1 = {}, acc2 = {}, acc3 = {};
  for (int k0 = 0; k0 < CCH; k0 += 32) {
    v16h bt = load_b_rm(Yb, (long)NSP, n0 + l, k0, half);   // streamed Y tile
    v16h a0 = load_a_rm(Mb, CCH, i0 +  0 + l, k0, half);
    v16h a1 = load_a_rm(Mb, CCH, i0 + 16 + l, k0, half);
    v16h a2 = load_a_rm(Mb, CCH, i0 + 32 + l, k0, half);
    v16h a3 = load_a_rm(Mb, CCH, i0 + 48 + l, k0, half);
    acc0 = WMMA_F16(a0, bt, acc0);
    acc1 = WMMA_F16(a1, bt, acc1);
    acc2 = WMMA_F16(a2, bt, acc2);
    acc3 = WMMA_F16(a3, bt, acc3);
  }

  float* Ob = out + (long)b * CCH * NSP;
  const float* cb = cvec + b * CCH;
#pragma unroll
  for (int r = 0; r < 8; ++r) {
    const int i = i0 + r + half * 8;
    Ob[(long)(i +  0) * NSP + n0 + l] = acc0[r] + cb[i +  0];
    Ob[(long)(i + 16) * NSP + n0 + l] = acc1[r] + cb[i + 16];
    Ob[(long)(i + 32) * NSP + n0 + l] = acc2[r] + cb[i + 32];
    Ob[(long)(i + 48) * NSP + n0 + l] = acc3[r] + cb[i + 48];
  }
}

__global__ void zero_kernel(float* __restrict__ p, int n) {
  const int i = blockIdx.x * blockDim.x + threadIdx.x;
  if (i < n) p[i] = 0.f;
}

// ---------------------------------------------------------------------------
extern "C" void kernel_launch(void* const* d_in, const int* in_sizes, int n_in,
                              void* d_out, int out_size, void* d_ws, size_t ws_size,
                              hipStream_t stream) {
  const float* x  = (const float*)d_in[0];
  const float* y  = (const float*)d_in[1];
  const float* Wq = (const float*)d_in[2];
  const float* bq = (const float*)d_in[3];
  const float* Wk = (const float*)d_in[4];
  const float* bk = (const float*)d_in[5];
  const float* Wv = (const float*)d_in[6];
  const float* bv = (const float*)d_in[7];
  float* out = (float*)d_out;
  float* ws  = (float*)d_ws;

  // workspace layout (floats)
  const long nMat = (long)NB * CCH * CCH;  // 262144
  const long nVec = (long)NB * CCH;        // 1024
  float* G  = ws;                  // accumulated (zero-init)
  float* sx = G  + nMat;           // accumulated (zero-init)
  float* sy = sx + nVec;           // accumulated (zero-init)
  float* u  = sy + nVec;
  float* w  = u  + nVec;
  float* T  = w  + nVec;
  float* S  = T  + nMat;           // scores -> softmax weights in place
  float* Mm = S  + nMat;
  float* cv = Mm + nMat;
  (void)ws_size; (void)in_sizes; (void)n_in; (void)out_size;

  const int zn = (int)(nMat + 2 * nVec);   // G, sx, sy
  zero_kernel<<<(zn + 255) / 256, 256, 0, stream>>>(G, zn);

  // pass 1: 4 batches x (8x8) 32x32-blocks x 32 K-slices = 8192 waves
  gram_kernel<<<(NB * 8 * 8 * KSLICES * 32) / 256, 256, 0, stream>>>(x, y, G, sx, sy);

  uv_kernel<<<NB, CCH, 0, stream>>>(Wq, Wk, sx, sy, u, w);

  // T = Wq * G       (A shared across batches, B per batch)
  smallgemm_kernel<<<(NB * 16 * 16 * 32) / 256, 256, 0, stream>>>(Wq, 0, G, nMat / NB, T);

  // S = T * Wk^T + rank-1 terms
  score_kernel<<<(NB * 16 * 16 * 32) / 256, 256, 0, stream>>>(T, Wk, u, w, bq, bk, S);

  // softmax rows + c = A bv
  softmax_kernel<<<NB * CCH, CCH, 0, stream>>>(S, bv, cv);

  // M = A * Wv       (A per batch, B shared)
  smallgemm_kernel<<<(NB * 16 * 16 * 32) / 256, 256, 0, stream>>>(S, nMat / NB, Wv, 0, Mm);

  // pass 2: out = M * Y + c; 4 x 2048 x 4 = 32768 waves
  fused_kernel<<<(NB * 4 * 2048 * 32) / 256, 256, 0, stream>>>(Mm, cv, y, out);
}
